// MHA_20899310862921
// MI455X (gfx1250) — matrix operand order repaired and checked by reading, not compiled, not run on hardware
//
#include <hip/hip_runtime.h>
#include <math.h>

// ---------------------------------------------------------------------------
// MHA forward for MI455X (gfx1250, wave32, WMMA).
// B=2, S=2048, D=1024, H=16, hd=64.
// f32->f16 convert -> fused QKV WMMA GEMM (async-LDS double-buffered B) ->
// causal flash attention (online softmax, register double-buffered K) ->
// output projection WMMA GEMM.
// All matmuls: V_WMMA_F32_16X16X32_F16 (f16 operands, fp32 accumulate).
// Weight tiles stream HBM->LDS with GLOBAL_LOAD_ASYNC_TO_LDS_B128 (ASYNCcnt),
// branchless 2-deep pipeline (prefetch address wraps mod DM so the loop body
// is uniform; dangling tail copies drain at S_ENDPGM).
// ---------------------------------------------------------------------------

typedef __attribute__((ext_vector_type(16))) _Float16 v16h;
typedef __attribute__((ext_vector_type(8)))  float    v8f;

union F16Frag { v16h v; uint4 u[2]; };
union H4      { _Float16 h[4]; uint2 u; };

#define WMMA16(a, b, c) \
  __builtin_amdgcn_wmma_f32_16x16x32_f16(false, (a), false, (b), (short)0, (c), false, false)

#define DM    1024      // d_model
#define SEQ   2048      // sequence length
#define NB    2         // batch
#define NH    16        // heads
#define HD    64        // head dim
#define ROWS  (NB*SEQ)  // 4096 GEMM rows
#define KSTEP 32
#define NSTEP (DM/KSTEP)

static __device__ __forceinline__ v8f vzero8() {
  v8f z = {0.f,0.f,0.f,0.f,0.f,0.f,0.f,0.f};
  return z;
}

// Async copy 16B global -> LDS (per lane), tracked by ASYNCcnt.
static __device__ __forceinline__ void async_g2l_b128(unsigned lds_byte_off,
                                                      const void* gsrc) {
  asm volatile("global_load_async_to_lds_b128 %0, %1, off"
               :: "v"(lds_byte_off), "v"(gsrc) : "memory");
}
static __device__ __forceinline__ void wait_async_le1() {
  asm volatile("s_wait_asynccnt 0x1" ::: "memory");
}
// Same-wave LDS ordering fence (single-wave blocks; no barrier needed).
static __device__ __forceinline__ void lds_fence() {
  asm volatile("s_wait_dscnt 0x0" ::: "memory");
}

// ---------------------------------------------------------------------- cvt
__global__ __launch_bounds__(256) void cvt_f32_f16_x4(const float4* __restrict__ src,
                                                      uint2* __restrict__ dst, int n4) {
  int i = blockIdx.x * blockDim.x + threadIdx.x;
  if (i < n4) {
    float4 f = src[i];
    H4 o;
    o.h[0] = (_Float16)f.x; o.h[1] = (_Float16)f.y;
    o.h[2] = (_Float16)f.z; o.h[3] = (_Float16)f.w;
    dst[i] = o.u;
  }
}

// ----------------------------------------------------------------- QKV GEMM
// out = x @ W^T for W in {Wq, Wk, Wv}.
// Block: 8 waves, tile 256(M) x 64(N); wave tile 32(M) x 64(N).
// B tile [64 n-rows x 32 k] (4 KB) double-buffered in LDS via async loads.
__global__ __launch_bounds__(256) void qkv_gemm(
    const _Float16* __restrict__ xh,
    const _Float16* __restrict__ wqh,
    const _Float16* __restrict__ wkh,
    const _Float16* __restrict__ wvh,
    _Float16* __restrict__ Qh,
    _Float16* __restrict__ Kh,
    _Float16* __restrict__ Vt) {
  __shared__ __align__(16) _Float16 Bs0[64 * KSTEP];
  __shared__ __align__(16) _Float16 Bs1[64 * KSTEP];

  const int tid  = threadIdx.x;
  const int lane = tid & 31;
  const int wv   = tid >> 5;
  const int hi   = lane >> 4;
  const int ln   = lane & 15;
  const int NT   = (3 * DM) / 64;                // 48 column tiles
  const int mblk = blockIdx.x / NT;
  const int ntile= blockIdx.x % NT;
  const int nbase= ntile * 64;
  const int w    = nbase >> 10;                  // 0=Q 1=K 2=V
  const int cbase= nbase & (DM - 1);
  const _Float16* __restrict__ W = (w == 0) ? wqh : (w == 1) ? wkh : wvh;
  const int wmbase = mblk * 256 + wv * 32;

  const int srow = tid >> 2;                     // 0..63
  const int schk = tid & 3;                      // 0..3
  const _Float16* sgp = W + (size_t)(cbase + srow) * DM + schk * 8;
  const unsigned sl0 = (unsigned)(uintptr_t)&Bs0[srow * KSTEP + schk * 8];
  const unsigned sl1 = (unsigned)(uintptr_t)&Bs1[srow * KSTEP + schk * 8];

  v8f acc[2][4];
  #pragma unroll
  for (int m = 0; m < 2; ++m)
    #pragma unroll
    for (int t = 0; t < 4; ++t) acc[m][t] = vzero8();

  const _Float16* arow0 = xh + (size_t)(wmbase + ln) * DM + hi * 16;
  const _Float16* arow1 = arow0 + (size_t)16 * DM;

  auto loadA = [&](int k0, F16Frag& a0, F16Frag& a1) {
    a0.u[0] = *(const uint4*)(arow0 + k0);
    a0.u[1] = *(const uint4*)(arow0 + k0 + 8);
    a1.u[0] = *(const uint4*)(arow1 + k0);
    a1.u[1] = *(const uint4*)(arow1 + k0 + 8);
  };
  auto mma = [&](const _Float16* bb, const F16Frag& a0, const F16Frag& a1) {
    F16Frag b[4];
    #pragma unroll
    for (int t = 0; t < 4; ++t) {                // distinct regs -> partial dscnt waits
      const _Float16* br = bb + (16 * t + ln) * KSTEP + hi * 16;
      b[t].u[0] = *(const uint4*)(br);
      b[t].u[1] = *(const uint4*)(br + 8);
    }
    #pragma unroll
    for (int t = 0; t < 4; ++t) {
      acc[0][t] = WMMA16(a0.v, b[t].v, acc[0][t]);
      acc[1][t] = WMMA16(a1.v, b[t].v, acc[1][t]);
    }
  };

  async_g2l_b128(sl0, (const void*)(sgp));           // buf0 <- k 0
  async_g2l_b128(sl1, (const void*)(sgp + KSTEP));   // buf1 <- k 32

  for (int s = 0; s < NSTEP; s += 2) {
    const int k0 = s * KSTEP;
    {
      F16Frag a0, a1;
      loadA(k0, a0, a1);
      wait_async_le1();                      // oldest (buf0) complete
      __syncthreads();
      mma(Bs0, a0, a1);
      __syncthreads();                       // all waves done with buf0
      async_g2l_b128(sl0, (const void*)(sgp + ((k0 + 2 * KSTEP) & (DM - 1))));
    }
    {
      F16Frag a0, a1;
      loadA(k0 + KSTEP, a0, a1);
      wait_async_le1();                      // oldest (buf1) complete
      __syncthreads();
      mma(Bs1, a0, a1);
      __syncthreads();
      async_g2l_b128(sl1, (const void*)(sgp + ((k0 + 3 * KSTEP) & (DM - 1))));
    }
  }

  const int h = cbase >> 6;
  #pragma unroll
  for (int m = 0; m < 2; ++m) {
    #pragma unroll
    for (int t = 0; t < 4; ++t) {
      const int d = 16 * t + ln;
      #pragma unroll
      for (int r = 0; r < 8; ++r) {
        const int row = wmbase + m * 16 + r + 8 * hi;
        const int sq  = row & (SEQ - 1);
        const int bb  = row >> 11;
        const float val = acc[m][t][r];
        if (w == 0)
          Qh[(((size_t)(bb * NH + h)) * SEQ + sq) * HD + d] = (_Float16)(val * 0.125f);
        else if (w == 1)
          Kh[(((size_t)(bb * NH + h)) * SEQ + sq) * HD + d] = (_Float16)val;
        else
          Vt[(((size_t)(bb * NH + h)) * HD + d) * SEQ + sq] = (_Float16)val;
      }
    }
  }
}

// ----------------------------------------------------------- flash attention
// One wave (block of 32) per (b, h, 16-query tile). Online softmax over
// 32-key chunks; causal mask only on the diagonal chunk. K fragments are
// double-buffered in registers; V fragments load before the softmax.
// Single-wave block: LDS P-bounce ordered with s_wait_dscnt, no barriers.
__global__ __launch_bounds__(32) void flash_attn(
    const _Float16* __restrict__ Qh,
    const _Float16* __restrict__ Kh,
    const _Float16* __restrict__ Vt,
    _Float16* __restrict__ AO) {
  __shared__ __align__(16) _Float16 P[16 * 32];

  const int lane = threadIdx.x & 31;
  const int hi   = lane >> 4;
  const int ln   = lane & 15;
  const int qt   = blockIdx.x & (SEQ / 16 - 1);
  const int bh   = blockIdx.x >> 7;
  const int qbase = qt * 16;

  const _Float16* __restrict__ Qb = Qh + (size_t)bh * SEQ * HD;
  const _Float16* __restrict__ Kb = Kh + (size_t)bh * SEQ * HD;
  const _Float16* __restrict__ Vb = Vt + (size_t)bh * HD * SEQ;

  F16Frag qa0, qa1;
  {
    const _Float16* qr = Qb + (size_t)(qbase + ln) * HD + hi * 16;
    qa0.u[0] = *(const uint4*)(qr);
    qa0.u[1] = *(const uint4*)(qr + 8);
    qa1.u[0] = *(const uint4*)(qr + 32);
    qa1.u[1] = *(const uint4*)(qr + 40);
  }

  float m[8], l[8];
  v8f o[4];
  #pragma unroll
  for (int r = 0; r < 8; ++r) { m[r] = -3.0e38f; l[r] = 0.f; }
  #pragma unroll
  for (int t = 0; t < 4; ++t) o[t] = vzero8();

  const int nchunks = (qt >> 1) + 1;

  auto loadK = [&](int kb, F16Frag* kf) {
    const _Float16* kr0 = Kb + (size_t)(kb + ln) * HD + hi * 16;
    const _Float16* kr1 = kr0 + (size_t)16 * HD;
    kf[0].u[0] = *(const uint4*)(kr0);      kf[0].u[1] = *(const uint4*)(kr0 + 8);
    kf[1].u[0] = *(const uint4*)(kr0 + 32); kf[1].u[1] = *(const uint4*)(kr0 + 40);
    kf[2].u[0] = *(const uint4*)(kr1);      kf[2].u[1] = *(const uint4*)(kr1 + 8);
    kf[3].u[0] = *(const uint4*)(kr1 + 32); kf[3].u[1] = *(const uint4*)(kr1 + 40);
  };

  auto processChunk = [&](int c, const F16Frag* kf) {
    const int kb = c * 32;

    // V fragments early (latency hides under score wmma + softmax)
    F16Frag vf[4];
    #pragma unroll
    for (int t2 = 0; t2 < 4; ++t2) {
      const _Float16* vr = Vb + (size_t)(16 * t2 + ln) * SEQ + kb + hi * 16;
      vf[t2].u[0] = *(const uint4*)(vr);
      vf[t2].u[1] = *(const uint4*)(vr + 8);
    }

    v8f s0 = vzero8(), s1 = vzero8();
    s0 = WMMA16(qa0.v, kf[0].v, s0);
    s0 = WMMA16(qa1.v, kf[1].v, s0);
    s1 = WMMA16(qa0.v, kf[2].v, s1);
    s1 = WMMA16(qa1.v, kf[3].v, s1);

    if (c == nchunks - 1) {                 // diagonal chunk: causal mask
      const int col0 = kb + ln, col1 = kb + 16 + ln;
      #pragma unroll
      for (int r = 0; r < 8; ++r) {
        const int q = qbase + r + 8 * hi;
        if (col0 > q) s0[r] = -3.0e38f;
        if (col1 > q) s1[r] = -3.0e38f;
      }
    }

    #pragma unroll
    for (int r = 0; r < 8; ++r) {
      float t = fmaxf(s0[r], s1[r]);
      #pragma unroll
      for (int off = 1; off < 16; off <<= 1) t = fmaxf(t, __shfl_xor(t, off));
      const float mnew  = fmaxf(m[r], t);
      const float alpha = __expf(m[r] - mnew);
      const float p0 = __expf(s0[r] - mnew);
      const float p1 = __expf(s1[r] - mnew);
      s0[r] = p0; s1[r] = p1;
      float rs = p0 + p1;
      #pragma unroll
      for (int off = 1; off < 16; off <<= 1) rs += __shfl_xor(rs, off);
      l[r] = l[r] * alpha + rs;
      m[r] = mnew;
      #pragma unroll
      for (int t2 = 0; t2 < 4; ++t2) o[t2][r] *= alpha;
    }

    // C-layout -> A-layout for P via LDS (same-wave: dscnt fence only)
    #pragma unroll
    for (int r = 0; r < 8; ++r) {
      const int row = r + 8 * hi;
      P[row * 32 + ln]      = (_Float16)s0[r];
      P[row * 32 + 16 + ln] = (_Float16)s1[r];
    }
    lds_fence();                            // stores visible before loads
    F16Frag pa;
    {
      const _Float16* pr = P + ln * 32 + hi * 16;
      pa.u[0] = *(const uint4*)(pr);
      pa.u[1] = *(const uint4*)(pr + 8);
    }
    lds_fence();                            // loads done before next overwrite

    #pragma unroll
    for (int t2 = 0; t2 < 4; ++t2) o[t2] = WMMA16(pa.v, vf[t2].v, o[t2]);
  };

  F16Frag kfA[4], kfB[4];
  loadK(0, kfA);
  int c = 0;
  while (c < nchunks) {
    if (c + 1 < nchunks) loadK((c + 1) * 32, kfB);
    processChunk(c, kfA);
    ++c;
    if (c >= nchunks) break;
    if (c + 1 < nchunks) loadK((c + 1) * 32, kfA);
    processChunk(c, kfB);
    ++c;
  }

  const int b = bh >> 4, h = bh & 15;
  #pragma unroll
  for (int r = 0; r < 8; ++r) {
    const float inv = 1.0f / l[r];
    const size_t rowg = (size_t)(b * SEQ + qbase + r + 8 * hi);
    #pragma unroll
    for (int t2 = 0; t2 < 4; ++t2)
      AO[rowg * DM + h * HD + 16 * t2 + ln] = (_Float16)(o[t2][r] * inv);
  }
}

// ----------------------------------------------------------- out projection
__global__ __launch_bounds__(256) void out_proj(
    const _Float16* __restrict__ AO,
    const _Float16* __restrict__ woh,
    float* __restrict__ out) {
  __shared__ __align__(16) _Float16 Bs0[64 * KSTEP];
  __shared__ __align__(16) _Float16 Bs1[64 * KSTEP];

  const int tid  = threadIdx.x;
  const int lane = tid & 31;
  const int wv   = tid >> 5;
  const int hi   = lane >> 4;
  const int ln   = lane & 15;
  const int NT   = DM / 64;                      // 16 column tiles
  const int mblk = blockIdx.x / NT;
  const int nbase= (blockIdx.x % NT) * 64;
  const int wmbase = mblk * 256 + wv * 32;

  const int srow = tid >> 2;
  const int schk = tid & 3;
  const _Float16* sgp = woh + (size_t)(nbase + srow) * DM + schk * 8;
  const unsigned sl0 = (unsigned)(uintptr_t)&Bs0[srow * KSTEP + schk * 8];
  const unsigned sl1 = (unsigned)(uintptr_t)&Bs1[srow * KSTEP + schk * 8];

  v8f acc[2][4];
  #pragma unroll
  for (int m = 0; m < 2; ++m)
    #pragma unroll
    for (int t = 0; t < 4; ++t) acc[m][t] = vzero8();

  const _Float16* arow0 = AO + (size_t)(wmbase + ln) * DM + hi * 16;
  const _Float16* arow1 = arow0 + (size_t)16 * DM;

  auto loadA = [&](int k0, F16Frag& a0, F16Frag& a1) {
    a0.u[0] = *(const uint4*)(arow0 + k0);
    a0.u[1] = *(const uint4*)(arow0 + k0 + 8);
    a1.u[0] = *(const uint4*)(arow1 + k0);
    a1.u[1] = *(const uint4*)(arow1 + k0 + 8);
  };
  auto mma = [&](const _Float16* bb, const F16Frag& a0, const F16Frag& a1) {
    F16Frag b[4];
    #pragma unroll
    for (int t = 0; t < 4; ++t) {
      const _Float16* br = bb + (16 * t + ln) * KSTEP + hi * 16;
      b[t].u[0] = *(const uint4*)(br);
      b[t].u[1] = *(const uint4*)(br + 8);
    }
    #pragma unroll
    for (int t = 0; t < 4; ++t) {
      acc[0][t] = WMMA16(a0.v, b[t].v, acc[0][t]);
      acc[1][t] = WMMA16(a1.v, b[t].v, acc[1][t]);
    }
  };

  async_g2l_b128(sl0, (const void*)(sgp));
  async_g2l_b128(sl1, (const void*)(sgp + KSTEP));

  for (int s = 0; s < NSTEP; s += 2) {
    const int k0 = s * KSTEP;
    {
      F16Frag a0, a1;
      loadA(k0, a0, a1);
      wait_async_le1();
      __syncthreads();
      mma(Bs0, a0, a1);
      __syncthreads();
      async_g2l_b128(sl0, (const void*)(sgp + ((k0 + 2 * KSTEP) & (DM - 1))));
    }
    {
      F16Frag a0, a1;
      loadA(k0 + KSTEP, a0, a1);
      wait_async_le1();
      __syncthreads();
      mma(Bs1, a0, a1);
      __syncthreads();
      async_g2l_b128(sl1, (const void*)(sgp + ((k0 + 3 * KSTEP) & (DM - 1))));
    }
  }

  #pragma unroll
  for (int m = 0; m < 2; ++m)
    #pragma unroll
    for (int t = 0; t < 4; ++t)
      #pragma unroll
      for (int r = 0; r < 8; ++r) {
        const size_t row = (size_t)(wmbase + m * 16 + r + 8 * hi);
        out[row * DM + nbase + 16 * t + ln] = acc[m][t][r];
      }
}

// ------------------------------------------------------------------- launch
extern "C" void kernel_launch(void* const* d_in, const int* in_sizes, int n_in,
                              void* d_out, int out_size, void* d_ws, size_t ws_size,
                              hipStream_t stream) {
  (void)in_sizes; (void)n_in; (void)out_size; (void)ws_size;
  const float* x  = (const float*)d_in[0];
  const float* wq = (const float*)d_in[1];
  const float* wk = (const float*)d_in[2];
  const float* wv = (const float*)d_in[3];
  const float* wo = (const float*)d_in[4];
  float* out = (float*)d_out;

  const size_t NX = (size_t)ROWS * DM;           // 4,194,304
  const size_t NW = (size_t)DM * DM;             // 1,048,576

  _Float16* p   = (_Float16*)d_ws;
  _Float16* xh  = p; p += NX;
  _Float16* wqh = p; p += NW;
  _Float16* wkh = p; p += NW;
  _Float16* wvh = p; p += NW;
  _Float16* woh = p; p += NW;
  _Float16* Qh  = p; p += NX;
  _Float16* Kh  = p; p += NX;
  _Float16* Vt  = p; p += NX;
  _Float16* AO  = p; p += NX;

  cvt_f32_f16_x4<<<(int)(NX / 4 / 256), 256, 0, stream>>>((const float4*)x,  (uint2*)xh,  (int)(NX / 4));
  cvt_f32_f16_x4<<<(int)(NW / 4 / 256), 256, 0, stream>>>((const float4*)wq, (uint2*)wqh, (int)(NW / 4));
  cvt_f32_f16_x4<<<(int)(NW / 4 / 256), 256, 0, stream>>>((const float4*)wk, (uint2*)wkh, (int)(NW / 4));
  cvt_f32_f16_x4<<<(int)(NW / 4 / 256), 256, 0, stream>>>((const float4*)wv, (uint2*)wvh, (int)(NW / 4));
  cvt_f32_f16_x4<<<(int)(NW / 4 / 256), 256, 0, stream>>>((const float4*)wo, (uint2*)woh, (int)(NW / 4));

  qkv_gemm<<<768, 256, 0, stream>>>(xh, wqh, wkh, wvh, Qh, Kh, Vt);
  flash_attn<<<4096, 32, 0, stream>>>(Qh, Kh, Vt, AO);
  out_proj<<<256, 256, 0, stream>>>(AO, woh, out);
}
